// CosineWeightedQuantization_79525614452871
// MI455X (gfx1250) — matrix operand order repaired and checked by reading, not compiled, not run on hardware
//
#include <hip/hip_runtime.h>
#include <hip/hip_bf16.h>
#include <stdint.h>

// CDNA5 / gfx1250 — wave32, WMMA 16x16x32 bf16, async global->LDS staging.
// Each wave owns 32 tokens (two 16-row A tiles) so every LDS B-fragment feeds
// two WMMAs: 1 ds_load_b128 per v_wmma, two independent accumulator chains.
typedef __attribute__((ext_vector_type(16))) __bf16 v16bf;
typedef __attribute__((ext_vector_type(8)))  __bf16 v8bf;
typedef __attribute__((ext_vector_type(8)))  float  v8f;

#define D_DIM        256
#define K_CODES      4096
#define THREADS      128                       // 4 waves
#define TOK_PER_BLK  128                       // 4 waves * 32 tokens
#define CHUNK_CODES  64
#define N_CHUNKS     (K_CODES / CHUNK_CODES)   // 64
#define N_TOKENS     65536

#if __has_builtin(__builtin_amdgcn_sched_barrier)
#define SCHED_FENCE() __builtin_amdgcn_sched_barrier(0)
#else
#define SCHED_FENCE() asm volatile("" ::: "memory")
#endif

__device__ __forceinline__ float dot4sq(float4 a) {
  return a.x*a.x + a.y*a.y + a.z*a.z + a.w*a.w;
}

// 16 contiguous bf16 from LDS -> v16bf (two ds_load_b128)
__device__ __forceinline__ v16bf ld_frag16(const __bf16* p) {
  v8bf lo = *(const v8bf*)p;
  v8bf hh = *(const v8bf*)(p + 8);
  v16bf r;
#pragma unroll
  for (int i = 0; i < 8; ++i) { r[i] = lo[i]; r[i + 8] = hh[i]; }
  return r;
}

// Async copy one 32 KB codebook chunk into LDS: 128 threads x 16 x b128.
// GVS mode: saddr = uniform chunk base (SGPR pair), vaddr = 32-bit lane offset,
// vdst = LDS byte address. Tracked by ASYNCcnt.
__device__ __forceinline__ void async_stage_chunk(uint32_t lds_base,
                                                  const __bf16* gsrc, int tid) {
  const uint64_t sbase = (uint64_t)(uintptr_t)gsrc;
  const uint32_t voff0 = (uint32_t)tid * 16u;
#pragma unroll
  for (int i = 0; i < 16; ++i) {
    uint32_t off = voff0 + (uint32_t)i * 2048u;
    asm volatile("global_load_async_to_lds_b128 %0, %1, %2"
                 :
                 : "v"(lds_base + off), "v"(off), "s"(sbase)
                 : "memory");
  }
}

__device__ __forceinline__ void wait_async0() {
  asm volatile("s_wait_asynccnt 0x0" ::: "memory");
}

__global__ void zero_loss_kernel(float* p) { *p = 0.0f; }

// Normalize codebook rows: bf16 unit rows + 1/max(norm,eps) table.
__global__ __launch_bounds__(256) void cbnorm_kernel(const float* __restrict__ cb,
                                                     __bf16* __restrict__ cbn,
                                                     float* __restrict__ invn) {
  int gw = (int)((blockIdx.x * blockDim.x + threadIdx.x) >> 5);
  int l  = (int)(threadIdx.x & 31);
  int nw = (int)((gridDim.x * blockDim.x) >> 5);
  for (int row = gw; row < K_CODES; row += nw) {
    const float4* p = (const float4*)(cb + (size_t)row * D_DIM);
    float4 a = p[l * 2 + 0];
    float4 b = p[l * 2 + 1];
    float s = dot4sq(a) + dot4sq(b);
#pragma unroll
    for (int off = 16; off >= 1; off >>= 1) s += __shfl_xor(s, off, 32);
    float inv = 1.0f / fmaxf(sqrtf(s), 1e-12f);
    v8bf v;
    v[0] = (__bf16)(a.x * inv); v[1] = (__bf16)(a.y * inv);
    v[2] = (__bf16)(a.z * inv); v[3] = (__bf16)(a.w * inv);
    v[4] = (__bf16)(b.x * inv); v[5] = (__bf16)(b.y * inv);
    v[6] = (__bf16)(b.z * inv); v[7] = (__bf16)(b.w * inv);
    *(v8bf*)(cbn + (size_t)row * D_DIM + l * 8) = v;
    if (l == 0) invn[row] = inv;
  }
}

// Build one 16-token A tile (raw x as bf16) + per-row inverse norm.
// unroll 2: keep at most 8 float4 loads in flight so the prologue stays under
// the 256-VGPR cap alongside the 128 VGPRs of finished A fragments (no spills).
__device__ __forceinline__ float build_atile(const float* xrow, int hi, v16bf* af) {
  float s = 0.0f;
#pragma unroll 2
  for (int kc = 0; kc < 8; ++kc) {
    const float4* p0 = (const float4*)(xrow + kc * 32 + hi * 16);
    float4 a0 = p0[0], a1 = p0[1];
    const float4* p1 = (const float4*)(xrow + kc * 32 + hi * 16 + 8);
    float4 b0 = p1[0], b1 = p1[1];
    s += dot4sq(a0) + dot4sq(a1) + dot4sq(b0) + dot4sq(b1);
    v16bf f;
    f[0]  = (__bf16)a0.x; f[1]  = (__bf16)a0.y; f[2]  = (__bf16)a0.z; f[3]  = (__bf16)a0.w;
    f[4]  = (__bf16)a1.x; f[5]  = (__bf16)a1.y; f[6]  = (__bf16)a1.z; f[7]  = (__bf16)a1.w;
    f[8]  = (__bf16)b0.x; f[9]  = (__bf16)b0.y; f[10] = (__bf16)b0.z; f[11] = (__bf16)b0.w;
    f[12] = (__bf16)b1.x; f[13] = (__bf16)b1.y; f[14] = (__bf16)b1.z; f[15] = (__bf16)b1.w;
    af[kc] = f;
  }
  s += __shfl_xor(s, 16, 32);                 // full-row sum of squares
  return 1.0f / fmaxf(sqrtf(s), 1e-12f);      // lane holds inv-norm of its row m
}

__global__ __launch_bounds__(THREADS)
__attribute__((amdgpu_num_vgpr(256)))
void vq_main_kernel(const float* __restrict__ x,
                    const float* __restrict__ cb,
                    const __bf16* __restrict__ cbn,
                    const float* __restrict__ invn,
                    float* __restrict__ out_ids,
                    float* __restrict__ out_emb,
                    float* __restrict__ out_loss) {
  // Double-buffered codebook tile: 2 x 32 KB = 64 KB LDS.
  __shared__ __align__(16) __bf16 cbs[2][CHUNK_CODES * D_DIM];

  const int tid = (int)threadIdx.x;
  const int w   = tid >> 5;        // wave 0..3, owns tokens [w*32, w*32+32)
  const int l   = tid & 31;
  const int m   = l & 15;          // A row / B column / D column
  const int hi  = l >> 4;          // lane half
  const int token0 = (int)blockIdx.x * TOK_PER_BLK;

  // LDS byte offsets of the two buffers (low 32 bits of generic ptr = LDS addr).
  const uint32_t lds_buf0 = (uint32_t)(uintptr_t)&cbs[0][0];
  const uint32_t lds_buf1 = (uint32_t)(uintptr_t)&cbs[1][0];

  // Kick off chunk 0 into buffer 0 before doing the A-tile work.
  async_stage_chunk(lds_buf0, cbn, tid);

  // ---- Two A tiles per wave (contiguous k = 16*hi + s layout), built sequentially. ----
  v16bf afragA[8], afragB[8];
  const float* xrow0 = x + (size_t)(token0 + w * 32 + m) * D_DIM;
  const float* xrow1 = x + (size_t)(token0 + w * 32 + 16 + m) * D_DIM;
  float inv_row0 = build_atile(xrow0, hi, afragA);
  SCHED_FENCE();
  float inv_row1 = build_atile(xrow1, hi, afragB);
  SCHED_FENCE();

  float best_v0[8], best_v1[8];
  int   best_i0[8], best_i1[8];
#pragma unroll
  for (int r = 0; r < 8; ++r) {
    best_v0[r] = -3.4e38f; best_i0[r] = 0;
    best_v1[r] = -3.4e38f; best_i1[r] = 0;
  }

  // ---- Sweep codebook with an async double-buffered pipeline. ----
  for (int ch = 0; ch < N_CHUNKS; ++ch) {
    const int cur = ch & 1;
    wait_async0();          // my chunk-ch asyncs have landed
    __syncthreads();        // whole block's chunk is visible
    if (ch + 1 < N_CHUNKS)  // overlap next fetch with compute
      async_stage_chunk(cur ? lds_buf0 : lds_buf1,
                        cbn + (size_t)(ch + 1) * CHUNK_CODES * D_DIM, tid);

    const __bf16* buf = &cbs[cur][0];
#pragma unroll
    for (int nt = 0; nt < 4; ++nt) {
      // B (32x16): lane column m holds K {k0+hi*16..+16} = contiguous 32B of
      // codebook row (nt*16+m). One fragment feeds BOTH token tiles.
      const __bf16* bp = buf + (size_t)(nt * 16 + m) * D_DIM + hi * 16;
      v8f c0 = {};
      v8f c1 = {};
#pragma unroll
      for (int kc = 0; kc < 8; ++kc) {
        v16bf bf = ld_frag16(bp + kc * 32);
        c0 = __builtin_amdgcn_wmma_f32_16x16x32_bf16(false, afragA[kc], false, bf,
                                                     (short)0, c0, false, false);
        c1 = __builtin_amdgcn_wmma_f32_16x16x32_bf16(false, afragB[kc], false, bf,
                                                     (short)0, c1, false, false);
      }
      const int code = ch * CHUNK_CODES + nt * 16 + m;
#pragma unroll
      for (int r = 0; r < 8; ++r) {
        float v0 = c0[r];
        if (v0 > best_v0[r]) { best_v0[r] = v0; best_i0[r] = code; }
        float v1 = c1[r];
        if (v1 > best_v1[r]) { best_v1[r] = v1; best_i1[r] = code; }
      }
    }
  }

  // ---- Argmax reduce across the 16 column-lanes of each half-wave. ----
#pragma unroll
  for (int r = 0; r < 8; ++r) {
    float v0 = best_v0[r]; int i0 = best_i0[r];
    float v1 = best_v1[r]; int i1 = best_i1[r];
#pragma unroll
    for (int off = 1; off < 16; off <<= 1) {
      float ov0 = __shfl_xor(v0, off, 32);
      int   oi0 = __shfl_xor(i0, off, 32);
      if (ov0 > v0 || (ov0 == v0 && oi0 < i0)) { v0 = ov0; i0 = oi0; }
      float ov1 = __shfl_xor(v1, off, 32);
      int   oi1 = __shfl_xor(i1, off, 32);
      if (ov1 > v1 || (ov1 == v1 && oi1 < i1)) { v1 = ov1; i1 = oi1; }
    }
    best_v0[r] = v0; best_i0[r] = i0;
    best_v1[r] = v1; best_i1[r] = i1;
  }

  // ---- Finalize both tiles: gather c_k, weight by cos_k, emit ids + loss. ----
  float lpart = 0.0f;
#pragma unroll
  for (int t = 0; t < 2; ++t) {
#pragma unroll
    for (int r = 0; r < 8; ++r) {
      const int lrow  = hi * 8 + r;                  // tile-local token row 0..15
      const int token = token0 + w * 32 + t * 16 + lrow;
      const int   id   = t ? best_i1[r] : best_i0[r];
      const float dotv = t ? best_v1[r] : best_v0[r]; // sum(x * cb_n[id])
      const float rinv = __shfl(t ? inv_row1 : inv_row0, lrow, 32);
      const float cosk = fmaxf(dotv, 1e-6f);
      const float scale = cosk * invn[id];           // emb = scale * cb[id]
      const float4* crow = (const float4*)(cb + (size_t)id * D_DIM);
      float4* orow = (float4*)(out_emb + (size_t)token * D_DIM);
#pragma unroll
      for (int j = 0; j < 4; ++j) {                  // 16 lanes x 16 floats = 256
        float4 vv = crow[m * 4 + j];
        vv.x *= scale; vv.y *= scale; vv.z *= scale; vv.w *= scale;
        orow[m * 4 + j] = vv;
      }
      if (m == 0) {
        out_ids[token] = (float)id;
        lpart += 1.0f - dotv * rinv;                 // 1 - cos_sim(x, c_k)
      }
    }
  }

  // Wave-local loss reduce (lanes 0 and 16 hold partials), one atomic per wave.
  lpart += __shfl_xor(lpart, 16, 32);
  if (l == 0) atomicAdd(out_loss, lpart * (1.25f / (float)N_TOKENS));
}

extern "C" void kernel_launch(void* const* d_in, const int* in_sizes, int n_in,
                              void* d_out, int out_size, void* d_ws, size_t ws_size,
                              hipStream_t stream) {
  const float* x  = (const float*)d_in[0];   // [64,1024,256] f32
  const float* cb = (const float*)d_in[1];   // [4096,256]    f32

  // Workspace: bf16 normalized codebook (2 MB) + inverse-norm table (16 KB).
  __bf16* cbn = (__bf16*)d_ws;
  float*  invn = (float*)((char*)d_ws + (size_t)K_CODES * D_DIM * sizeof(__bf16));

  float* out      = (float*)d_out;
  float* out_ids  = out;                                   // 65536 (ids as f32 values)
  float* out_emb  = out + N_TOKENS;                        // 65536*256
  float* out_loss = out + N_TOKENS + (size_t)N_TOKENS * D_DIM;  // 1

  zero_loss_kernel<<<1, 1, 0, stream>>>(out_loss);
  cbnorm_kernel<<<64, 256, 0, stream>>>(cb, cbn, invn);
  vq_main_kernel<<<N_TOKENS / TOK_PER_BLK, THREADS, 0, stream>>>(x, cb, cbn, invn,
                                                                 out_ids, out_emb,
                                                                 out_loss);
}
